// DecoderCenter_23519240913094
// MI455X (gfx1250) — compile-verified, matrix operand order checked by reading, not run on hardware
//
#include <hip/hip_runtime.h>
#include <hip/hip_bf16.h>

typedef __attribute__((ext_vector_type(16))) _Float16 v16h;
typedef __attribute__((ext_vector_type(8)))  _Float16 v8h;
typedef __attribute__((ext_vector_type(8)))  float    v8f;

// ---------------------------------------------------------------------------
// CDNA5 async-copy helpers (GLOBAL_LOAD_ASYNC_TO_LDS_B128, ASYNCcnt)
// Generic pointers to LDS carry the LDS byte offset in their low 32 bits
// (ISA 10.2), so truncation yields the DS address.
// ---------------------------------------------------------------------------
__device__ __forceinline__ void async_copy_b128(void* lds_ptr, const void* gptr) {
    unsigned lds_off = (unsigned)(unsigned long long)lds_ptr;
    asm volatile("global_load_async_to_lds_b128 %0, %1, off"
                 :: "v"(lds_off), "v"(gptr) : "memory");
}
__device__ __forceinline__ void wait_async0() {
    asm volatile("s_wait_asynccnt 0x0" ::: "memory");
}

// ---------------------------------------------------------------------------
// Small helper kernels
// ---------------------------------------------------------------------------

__global__ void cvt_f32_to_f16(const float* __restrict__ s, _Float16* __restrict__ d, int n) {
    int i = blockIdx.x * blockDim.x + threadIdx.x;
    if (i < n) d[i] = (_Float16)s[i];
}

// Positional encoding: pe[row][oct*2+axis] = sin(points[row][axis] * 2^(oct/2))
__global__ void pe_kernel(const float* __restrict__ pts, _Float16* __restrict__ pe, int total) {
    int i = blockIdx.x * blockDim.x + threadIdx.x;
    if (i >= total) return;
    int col = i & 31;
    int row = i >> 5;
    int oct = col >> 1, axis = col & 1;
    float f = exp2f(0.5f * (float)oct);
    pe[i] = (_Float16)__sinf(pts[row * 2 + axis] * f);
}

// Direct grouped 3x3 conv + LeakyReLU (conv stack is <2% of total FLOPs)
__global__ void conv3x3_leaky(const float* __restrict__ x, const float* __restrict__ w,
                              const float* __restrict__ bias, float* __restrict__ y,
                              int B, int Cin, int H, int Wd, int OC, int OH, int OW,
                              int groups, int pad) {
    int idx = blockIdx.x * blockDim.x + threadIdx.x;
    int total = B * OC * OH * OW;
    if (idx >= total) return;
    int ow = idx % OW; int t = idx / OW;
    int oh = t % OH;   t /= OH;
    int oc = t % OC;   int b = t / OC;
    int icpg = Cin / groups;
    int ocpg = OC / groups;
    int g   = oc / ocpg;
    int ic0 = g * icpg;
    float acc = bias[oc];
    const float* wp = w + (size_t)oc * icpg * 9;
    for (int ic = 0; ic < icpg; ++ic) {
        const float* xp = x + ((size_t)b * Cin + ic0 + ic) * H * Wd;
        const float* wk = wp + ic * 9;
        #pragma unroll
        for (int kh = 0; kh < 3; ++kh) {
            int ih = oh - pad + kh;
            if (ih < 0 || ih >= H) continue;
            #pragma unroll
            for (int kw = 0; kw < 3; ++kw) {
                int iw = ow - pad + kw;
                if (iw < 0 || iw >= Wd) continue;
                acc = fmaf(xp[ih * Wd + iw], wk[kh * 3 + kw], acc);
            }
        }
    }
    y[idx] = acc >= 0.f ? acc : 0.01f * acc;
}

// Final 512->3 projection + sigmoid + affine, transposed write (B,3,N). Tiny.
__global__ void out_proj(const _Float16* __restrict__ R, const float* __restrict__ Wo,
                         const float* __restrict__ bo, float* __restrict__ out, int M) {
    int i = blockIdx.x * blockDim.x + threadIdx.x;
    if (i >= M * 3) return;
    int c = i % 3, row = i / 3;
    float acc = bo[c];
    const _Float16* rp = R + (size_t)row * 512;
    const float* wp = Wo + c * 512;
    for (int k = 0; k < 512; ++k) acc = fmaf((float)rp[k], wp[k], acc);
    float s = 1.f / (1.f + expf(-acc));
    int b = row >> 13, n = row & 8191;
    out[((size_t)b * 3 + c) * 8192 + n] = s * 1.1f - 0.05f;
}

// ---------------------------------------------------------------------------
// WMMA GEMM:  out = leaky(A @ W^T + bias) [+ resid]   (f16 in, f32 acc, f16 out)
//   A is a virtual concat: cols [0,k0) from A0, cols [k0,K) from A1.
//   Row of A0/A1 is (globalRow >> shN)  (sh=13 implements feat repeat).
//   W is (N_out, K) row-major f16 -> B fragments address identically to A.
// Block: 256 thr = 8 waves, tile 64(M) x 128(N), K-step 32.
// Wave(wm 0..1, wn 0..3) owns 32x32 -> 2x2 v_wmma_f32_16x16x32_f16 per K-step.
// Tiles staged with GLOBAL_LOAD_ASYNC_TO_LDS_B128 into ping-pong LDS buffers:
// per thread per tile: 1 async B128 for A (64x32h = 256 chunks) and
// 2 async B128 for W (128x32h = 512 chunks).  K is always a multiple of 32,
// N of 128, M of 64 -> no guards, EXEC all-ones around WMMA as required.
// ---------------------------------------------------------------------------

#define GTM 64
#define GTN 128
#define GKT 32
#define LPAD 40   // halves per LDS row: 80B stride, 16B aligned, conflict-free

__global__ __launch_bounds__(256) void gemm_leaky_wmma(
    const _Float16* __restrict__ A0, int ld0, int sh0, int k0,
    const _Float16* __restrict__ A1, int ld1, int sh1,
    const _Float16* __restrict__ Wt, int ldw,
    const float* __restrict__ bias,
    const _Float16* __restrict__ resid, int ldr,
    _Float16* __restrict__ out, int ldo,
    int K) {
    __shared__ _Float16 sA[2][GTM * LPAD];
    __shared__ _Float16 sB[2][GTN * LPAD];

    const int tid  = threadIdx.x;
    const int lane = tid & 31;
    const int wave = tid >> 5;
    const int wm   = wave & 1;   // 2 waves along M
    const int wn   = wave >> 1;  // 4 waves along N
    const int bm   = blockIdx.x * GTM;
    const int bn   = blockIdx.y * GTN;

    // async staging coordinates (16B chunks)
    const int ar  = tid >> 2;          // A row 0..63
    const int ac  = (tid & 3) * 8;     // A col in halves (chunk of 8)
    const int wr0 = tid >> 2;          // W rows 0..63 / 64..127
    const int wc  = (tid & 3) * 8;

    v8f acc[2][2] = {};

    auto issue_tile = [&](int kb, int buf) {
        const _Float16* srcA; int lda, sha, cb;
        if (kb < k0) { srcA = A0; lda = ld0; sha = sh0; cb = kb; }
        else         { srcA = A1; lda = ld1; sha = sh1; cb = kb - k0; }
        async_copy_b128(&sA[buf][ar * LPAD + ac],
                        srcA + (size_t)(unsigned)(((bm + ar) >> sha) * lda + cb + ac));
        async_copy_b128(&sB[buf][wr0 * LPAD + wc],
                        Wt + (size_t)(unsigned)((bn + wr0) * ldw + kb + wc));
        async_copy_b128(&sB[buf][(wr0 + 64) * LPAD + wc],
                        Wt + (size_t)(unsigned)((bn + wr0 + 64) * ldw + kb + wc));
    };

    // prologue: fill buffer 0
    issue_tile(0, 0);
    wait_async0();
    __syncthreads();

    const int l16 = lane & 15;
    const int kh  = (lane >> 4) * 8;
    const int nk  = K / GKT;
    int p = 0;
    for (int kt = 0; kt < nk; ++kt) {
        if (kt + 1 < nk) issue_tile((kt + 1) * GKT, p ^ 1);

        // ---- fragments per ISA 16-bit A/B layout ----
        // lane 0-15: K 0-7 in halves[0..7], K 16-23 in halves[8..15]
        // lane 16-31: K 8-15 / 24-31
        v16h af[2], bfv[2];
        #pragma unroll
        for (int t = 0; t < 2; ++t) {
            const _Float16* pa = &sA[p][(wm * 32 + t * 16 + l16) * LPAD];
            v8h alo = *reinterpret_cast<const v8h*>(pa + kh);
            v8h ahi = *reinterpret_cast<const v8h*>(pa + kh + 16);
            const _Float16* pb = &sB[p][(wn * 32 + t * 16 + l16) * LPAD];
            v8h blo = *reinterpret_cast<const v8h*>(pb + kh);
            v8h bhi = *reinterpret_cast<const v8h*>(pb + kh + 16);
            v16h fa, fb;
            #pragma unroll
            for (int j = 0; j < 8; ++j) {
                fa[j] = alo[j]; fa[8 + j] = ahi[j];
                fb[j] = blo[j]; fb[8 + j] = bhi[j];
            }
            af[t] = fa; bfv[t] = fb;
        }

        #pragma unroll
        for (int tm = 0; tm < 2; ++tm)
            #pragma unroll
            for (int tn = 0; tn < 2; ++tn)
                acc[tm][tn] = __builtin_amdgcn_wmma_f32_16x16x32_f16(
                    false, af[tm], false, bfv[tn], (short)0, acc[tm][tn],
                    false, false);

        wait_async0();      // tile kt+1 fully in LDS (this wave's issues)
        __syncthreads();    // publish across waves; also fences reads of buf p
        p ^= 1;
    }

    // ---- epilogue: bias + leaky (+ residual), uniform branch, 32-bit addr ----
    // C/D layout: lane 0-15 -> N=lane, M=v;  lane 16-31 -> N=lane-16, M=v+8
    const int rh = (lane >> 4) * 8;
    #pragma unroll
    for (int tm = 0; tm < 2; ++tm) {
        #pragma unroll
        for (int tn = 0; tn < 2; ++tn) {
            int col  = bn + wn * 32 + tn * 16 + l16;
            int row0 = bm + wm * 32 + tm * 16 + rh;
            float bv = bias[col];
            if (resid) {
                int ro = row0 * ldr + col;
                int oo = row0 * ldo + col;
                #pragma unroll
                for (int v = 0; v < 8; ++v) {
                    float xv = acc[tm][tn][v] + bv;
                    xv = xv >= 0.f ? xv : 0.01f * xv;
                    xv += (float)resid[(size_t)(unsigned)ro];
                    out[(size_t)(unsigned)oo] = (_Float16)xv;
                    ro += ldr; oo += ldo;
                }
            } else {
                int oo = row0 * ldo + col;
                #pragma unroll
                for (int v = 0; v < 8; ++v) {
                    float xv = acc[tm][tn][v] + bv;
                    xv = xv >= 0.f ? xv : 0.01f * xv;
                    out[(size_t)(unsigned)oo] = (_Float16)xv;
                    oo += ldo;
                }
            }
        }
    }
}

// ---------------------------------------------------------------------------
// Launcher
// ---------------------------------------------------------------------------

static inline size_t al256(size_t x) { return (x + 255) & ~(size_t)255; }
static inline int cdiv(int a, int b) { return (a + b - 1) / b; }

extern "C" void kernel_launch(void* const* d_in, const int* in_sizes, int n_in,
                              void* d_out, int out_size, void* d_ws, size_t ws_size,
                              hipStream_t stream) {
    const float* feature = (const float*)d_in[0];
    const float* points  = (const float*)d_in[1];
    const float* cw1 = (const float*)d_in[2];  const float* cb1 = (const float*)d_in[3];
    const float* cw2 = (const float*)d_in[4];  const float* cb2 = (const float*)d_in[5];
    const float* cw3 = (const float*)d_in[6];  const float* cb3 = (const float*)d_in[7];
    const float* cw4 = (const float*)d_in[8];  const float* cb4 = (const float*)d_in[9];
    const float* cw5 = (const float*)d_in[10]; const float* cb5 = (const float*)d_in[11];
    const float* lin1_W = (const float*)d_in[12]; const float* lin1_b = (const float*)d_in[13];
    const float* m1_Wf  = (const float*)d_in[14]; const float* m1_bf  = (const float*)d_in[15];
    const float* m1_Wg  = (const float*)d_in[16]; const float* m1_bg  = (const float*)d_in[17];
    const float* lin2_W = (const float*)d_in[18]; const float* lin2_b = (const float*)d_in[19];
    const float* m2_Wf  = (const float*)d_in[20]; const float* m2_bf  = (const float*)d_in[21];
    const float* m2_Wg  = (const float*)d_in[22]; const float* m2_bg  = (const float*)d_in[23];
    const float* out_W  = (const float*)d_in[24]; const float* out_b  = (const float*)d_in[25];

    const int M = 32768;  // B*N = 4*8192

    // workspace carve
    char* ws = (char*)d_ws;
    size_t off = 0;
    auto carve = [&](size_t bytes) { void* p = ws + off; off = al256(off + bytes); return p; };
    _Float16* pe     = (_Float16*)carve((size_t)M * 32 * 2);
    _Float16* feat   = (_Float16*)carve((size_t)4 * 2048 * 2);
    _Float16* Hb     = (_Float16*)carve((size_t)M * 1024 * 2);
    _Float16* Rb     = (_Float16*)carve((size_t)M * 512 * 2);
    _Float16* w_lin1 = (_Float16*)carve((size_t)1024 * 2080 * 2);
    _Float16* w_m1f  = (_Float16*)carve((size_t)8 * 512 * 512 * 2);
    _Float16* w_m1g  = (_Float16*)carve((size_t)8 * 512 * 512 * 2);
    _Float16* w_lin2 = (_Float16*)carve((size_t)512 * 1056 * 2);
    _Float16* w_m2f  = (_Float16*)carve((size_t)8 * 256 * 256 * 2);
    _Float16* w_m2g  = (_Float16*)carve((size_t)8 * 256 * 256 * 2);
    float*    cA     = (float*)carve((size_t)4 * 768 * 8 * 8 * 4);
    float*    cB     = (float*)carve((size_t)4 * 768 * 8 * 8 * 4);
    float*    c5     = (float*)carve((size_t)4 * 128 * 4 * 4 * 4);

    // ---- conv stack (f32, tiny) ----
    conv3x3_leaky<<<cdiv(4*768*8*8,256),256,0,stream>>>(feature, cw1, cb1, cA, 4,1792,8,8, 768,8,8, 4,1);
    conv3x3_leaky<<<cdiv(4*768*8*8,256),256,0,stream>>>(cA, cw2, cb2, cB, 4,768,8,8, 768,8,8, 3,1);
    conv3x3_leaky<<<cdiv(4*768*8*8,256),256,0,stream>>>(cB, cw3, cb3, cA, 4,768,8,8, 768,8,8, 2,1);
    conv3x3_leaky<<<cdiv(4*768*6*6,256),256,0,stream>>>(cA, cw4, cb4, cB, 4,768,8,8, 768,6,6, 3,0);
    conv3x3_leaky<<<cdiv(4*128*4*4,256),256,0,stream>>>(cB, cw5, cb5, c5, 4,768,6,6, 128,4,4, 1,0);
    cvt_f32_to_f16<<<cdiv(8192,256),256,0,stream>>>(c5, feat, 8192);

    // ---- positional encoding ----
    pe_kernel<<<cdiv(M*32,256),256,0,stream>>>(points, pe, M*32);

    // ---- weight f32 -> f16 ----
    cvt_f32_to_f16<<<cdiv(1024*2080,256),256,0,stream>>>(lin1_W, w_lin1, 1024*2080);
    cvt_f32_to_f16<<<cdiv(8*512*512,256),256,0,stream>>>(m1_Wf, w_m1f, 8*512*512);
    cvt_f32_to_f16<<<cdiv(8*512*512,256),256,0,stream>>>(m1_Wg, w_m1g, 8*512*512);
    cvt_f32_to_f16<<<cdiv(512*1056,256),256,0,stream>>>(lin2_W, w_lin2, 512*1056);
    cvt_f32_to_f16<<<cdiv(8*256*256,256),256,0,stream>>>(m2_Wf, w_m2f, 8*256*256);
    cvt_f32_to_f16<<<cdiv(8*256*256,256),256,0,stream>>>(m2_Wg, w_m2g, 8*256*256);

    // ---- lin1: h = leaky([pe | feat-repeat] @ lin1_W^T + b)  (32768x1024) ----
    {
        dim3 g(M / GTM, 1024 / GTN);
        gemm_leaky_wmma<<<g,256,0,stream>>>(pe, 32, 0, 32, feat, 2048, 13,
                                            w_lin1, 2080, lin1_b,
                                            nullptr, 0, Hb, 1024, 2080);
    }
    // ---- 8 additive couplings on 512-halves, in place on Hb ----
    {
        dim3 g(M / GTM, 512 / GTN);
        for (int i = 0; i < 8; ++i) {
            // y1 = h1 + leaky(h2 @ Wf^T + bf)  -> cols [0,512)
            gemm_leaky_wmma<<<g,256,0,stream>>>(Hb + 512, 1024, 0, 512, nullptr, 0, 0,
                                                w_m1f + (size_t)i*512*512, 512, m1_bf + i*512,
                                                Hb, 1024, Hb, 1024, 512);
            // y2 = h2 + leaky(y1 @ Wg^T + bg)  -> cols [512,1024)
            gemm_leaky_wmma<<<g,256,0,stream>>>(Hb, 1024, 0, 512, nullptr, 0, 0,
                                                w_m1g + (size_t)i*512*512, 512, m1_bg + i*512,
                                                Hb + 512, 1024, Hb + 512, 1024, 512);
        }
    }
    // ---- lin2: r = leaky([pe | h] @ lin2_W^T + b)  (32768x512) ----
    {
        dim3 g(M / GTM, 512 / GTN);
        gemm_leaky_wmma<<<g,256,0,stream>>>(pe, 32, 0, 32, Hb, 1024, 0,
                                            w_lin2, 1056, lin2_b,
                                            nullptr, 0, Rb, 512, 1056);
    }
    // ---- 8 additive couplings on 256-halves, in place on Rb ----
    {
        dim3 g(M / GTM, 256 / GTN);
        for (int i = 0; i < 8; ++i) {
            gemm_leaky_wmma<<<g,256,0,stream>>>(Rb + 256, 512, 0, 256, nullptr, 0, 0,
                                                w_m2f + (size_t)i*256*256, 256, m2_bf + i*256,
                                                Rb, 512, Rb, 512, 256);
            gemm_leaky_wmma<<<g,256,0,stream>>>(Rb, 512, 0, 256, nullptr, 0, 0,
                                                w_m2g + (size_t)i*256*256, 256, m2_bg + i*256,
                                                Rb + 256, 512, Rb + 256, 512, 256);
        }
    }
    // ---- output projection + sigmoid + affine, transposed to (B,3,N) ----
    out_proj<<<cdiv(M*3,256),256,0,stream>>>(Rb, out_W, out_b, (float*)d_out, M);
}
